// NATTEN_20761871909021
// MI455X (gfx1250) — matrix operand order repaired and checked by reading, not compiled, not run on hardware
//
#include <hip/hip_runtime.h>
#include <hip/hip_bf16.h>

typedef __attribute__((ext_vector_type(16))) _Float16 v16h;
typedef __attribute__((ext_vector_type(8)))  _Float16 v8h;
typedef __attribute__((ext_vector_type(4)))  _Float16 v4h;
typedef __attribute__((ext_vector_type(8)))  float    v8f;

#define Bn   2
#define Hh   192
#define Ww   88
#define FIN  133
#define FINP 160                 // FIN padded to 5*32 (f16 storage, pads = 0)
#define Cc   24
#define CcP  32                  // C padded to 32 (f16 storage, pads = 0)
#define NH   4
#define DH   6
#define DHP  8                   // dh padded to 8 (f16 storage, pads = 0)
#define WIN  25
#define HW   (Hh * Ww)           // 16896
#define BHW  (Bn * Hh * Ww)      // 33792
#define ZHW  (Bn * NH * HW)      // 135168 (per-head positions)
#define QSC  0.4082482904638631f // 1/sqrt(6)

// weight packs: f16, WMMA-B register order: [tile][lane][half]
#define TILE_H   512             // 32 lanes * 16 halves
#define WL_TILES 10              // 5 K-steps x 2 N-tiles
#define WQ_TILES 5               // 1 K-step  x 5 N-tiles
#define WP_TILES 2               // 1 K-step  x 2 N-tiles

// A (16xK, f16): lane L holds row m=L%16; half t -> k = a16_k(t, L/16)
//   t<8: k = t + 8*hi ; t>=8: k = 16 + (t-8) + 8*hi   (two 8-half runs)
__device__ __forceinline__ int a16_k(int t, int hi) {
  int g = t >> 1, r = t & 1;
  return ((g < 4) ? 0 : 16) + 2 * (g & 3) + r + 8 * hi;
}
// B packs (Kx16, f16): lane L holds col n=L%16; half t -> k = t + 16*(L/16), as a_from_padded's A
// C/D (16x16 f32): VGPR g -> (m = g + 8*(L/16), n = L%16)

__device__ __forceinline__ int iclampi(int v, int lo, int hi) {
  return v < lo ? lo : (v > hi ? hi : v);
}

// ---- A fragment from a padded-f16 row: exactly two b128 loads -------------
__device__ __forceinline__ v16h a_from_padded(const _Float16* row, int hi) {
  v8h r0 = *(const v8h*)(row + 16 * hi);
  v8h r1 = *(const v8h*)(row + 16 * hi + 8);
  v16h A;
#pragma unroll
  for (int t = 0; t < 8; ++t)  A[t] = r0[t];
#pragma unroll
  for (int t = 8; t < 16; ++t) A[t] = r1[t - 8];
  return A;
}

// ===========================================================================
// Kernel 0a: pack W_lin / W_qkv / W_proj into f16 WMMA-B register order.
// ===========================================================================
__global__ __launch_bounds__(256) void pack_weights_kernel(
    const float* __restrict__ Wl, const float* __restrict__ Wq,
    const float* __restrict__ Wp, _Float16* __restrict__ wlp,
    _Float16* __restrict__ wqp, _Float16* __restrict__ wpp) {
  int id = blockIdx.x * 256 + threadIdx.x;
  if (id < WL_TILES * TILE_H) {
    int tile = id / TILE_H, rem = id % TILE_H;
    int ln = rem >> 4, t = rem & 15;
    int s = tile >> 1, nt = tile & 1;
    int k = s * 32 + t + 16 * (ln >> 4), n = nt * 16 + (ln & 15);
    wlp[id] = (k < FIN && n < Cc) ? (_Float16)Wl[k * Cc + n] : (_Float16)0.0f;
  }
  if (id < WQ_TILES * TILE_H) {
    int tile = id / TILE_H, rem = id % TILE_H;
    int ln = rem >> 4, t = rem & 15;
    int k = t + 16 * (ln >> 4), n = tile * 16 + (ln & 15);
    wqp[id] = (k < Cc && n < 3 * Cc) ? (_Float16)Wq[k * (3 * Cc) + n]
                                     : (_Float16)0.0f;
  }
  if (id < WP_TILES * TILE_H) {
    int tile = id / TILE_H, rem = id % TILE_H;
    int ln = rem >> 4, t = rem & 15;
    int k = t + 16 * (ln >> 4), n = tile * 16 + (ln & 15);
    wpp[id] = (k < Cc && n < Cc) ? (_Float16)Wp[k * Cc + n] : (_Float16)0.0f;
  }
}

// ===========================================================================
// Kernel 0b: convert x -> f16 [BHW, 160], zero-padding cols >= 133.
// One block per row, 160 threads (5 waves).
// ===========================================================================
__global__ __launch_bounds__(FINP) void convert_x_kernel(
    const float* __restrict__ x, _Float16* __restrict__ x16) {
  int row = blockIdx.x, col = threadIdx.x;
  x16[row * FINP + col] =
      (col < FIN) ? (_Float16)x[row * FIN + col] : (_Float16)0.0f;
}

// ===========================================================================
// Kernel 0c: zero the K-pads of q/k/v (d=6,7) and of hb16/ao16 (cols 24..31).
// ===========================================================================
__global__ __launch_bounds__(256) void zero_pad_kernel(
    unsigned* __restrict__ qp, unsigned* __restrict__ kp,
    unsigned* __restrict__ vp, unsigned* __restrict__ hp,
    unsigned* __restrict__ ap) {
  int id = blockIdx.x * 256 + threadIdx.x;
  if (id < ZHW) {
    qp[id * 4 + 3] = 0u;  // halves 6,7 of the 8-half record
    kp[id * 4 + 3] = 0u;
    vp[id * 4 + 3] = 0u;
  }
  if (id < BHW) {
#pragma unroll
    for (int w = 12; w < 16; ++w) {  // halves 24..31 of the 32-half row
      hp[id * 16 + w] = 0u;
      ap[id * 16 + w] = 0u;
    }
  }
}

// ===========================================================================
// Kernel 1: h = relu(x @ W_lin + b_lin) -> f16 [BHW, 32]
// ===========================================================================
__global__ __launch_bounds__(32) void lin_relu_kernel(
    const _Float16* __restrict__ x16, const _Float16* __restrict__ wlp,
    const float* __restrict__ bl, _Float16* __restrict__ hb) {
  const int lane = threadIdx.x, hi = lane >> 4, lo = lane & 15;
  const int r0 = blockIdx.x * 16;
  const _Float16* xrow = x16 + (r0 + lo) * FINP;

  v16h A[5];
#pragma unroll
  for (int s = 0; s < 5; ++s) A[s] = a_from_padded(xrow + s * 32, hi);

#pragma unroll
  for (int nt = 0; nt < 2; ++nt) {
    v8f acc = {};
#pragma unroll
    for (int s = 0; s < 5; ++s) {
      v16h Bv = *(const v16h*)(wlp + ((s * 2 + nt) * TILE_H + lane * 16));
      acc = __builtin_amdgcn_wmma_f32_16x16x32_f16(false, A[s], false, Bv,
                                                   (short)0, acc, false, false);
    }
    int n = nt * 16 + lo;
    float bias = bl[n < Cc ? n : Cc - 1];
    if (n < Cc) {
#pragma unroll
      for (int g = 0; g < 8; ++g) {
        int m = g + 8 * hi;
        float v = acc[g] + bias;
        hb[(r0 + m) * CcP + n] = (_Float16)(v > 0.0f ? v : 0.0f);
      }
    }
  }
}

// ===========================================================================
// Kernel 2: qkv = h @ W_qkv + b_qkv -> f16 q(scaled)/k/v [B*NH, H, W, DHP]
// ===========================================================================
__global__ __launch_bounds__(32) void qkv_kernel(
    const _Float16* __restrict__ hb, const _Float16* __restrict__ wqp,
    const float* __restrict__ bq, _Float16* __restrict__ qb,
    _Float16* __restrict__ kb, _Float16* __restrict__ vb) {
  const int lane = threadIdx.x, hi = lane >> 4, lo = lane & 15;
  const int r0 = blockIdx.x * 16;

  v16h A = a_from_padded(hb + (r0 + lo) * CcP, hi);

#pragma unroll
  for (int nt = 0; nt < 5; ++nt) {
    v16h Bv = *(const v16h*)(wqp + (nt * TILE_H + lane * 16));
    v8f acc = {};
    acc = __builtin_amdgcn_wmma_f32_16x16x32_f16(false, A, false, Bv, (short)0,
                                                 acc, false, false);
    int n = nt * 16 + lo;
    float bias = bq[n < 3 * Cc ? n : 3 * Cc - 1];
    if (n < 3 * Cc) {
      int which = n / Cc, cc = n % Cc, head = cc / DH, d = cc % DH;
#pragma unroll
      for (int g = 0; g < 8; ++g) {
        int m = g + 8 * hi;
        int r = r0 + m;
        int bI = r / HW, hw = r % HW;
        int idx = ((bI * NH + head) * HW + hw) * DHP + d;
        float v = acc[g] + bias;
        if (which == 0)      qb[idx] = (_Float16)(v * QSC);
        else if (which == 1) kb[idx] = (_Float16)v;
        else                 vb[idx] = (_Float16)v;
      }
    }
  }
}

// ===========================================================================
// Kernel 3: neighborhood attention. grid=(6 jtiles, 192 rows, B*NH), 1 wave.
// ===========================================================================
__global__ __launch_bounds__(32) void natten_kernel(
    const _Float16* __restrict__ q, const _Float16* __restrict__ kg,
    const _Float16* __restrict__ vg, const float* __restrict__ rpb,
    _Float16* __restrict__ ao) {
  const int lane = threadIdx.x, hi = lane >> 4, lo = lane & 15;
  const int jt = blockIdx.x, i = blockIdx.y, z = blockIdx.z;
  const int head = z & (NH - 1);
  const int j0 = jt * 16;
  const int si = iclampi(i - WIN / 2, 0, Hh - WIN);
  const int bi = si - i + WIN - 1;
  const int u0 = iclampi(j0 - WIN / 2, 0, Ww - WIN);

  __shared__ float    sc[16][WIN * WIN + 7];      // f32 scores; col 625 dump
  __shared__ _Float16 af[16][WIN * WIN + 7];      // f16 normalized attention
  __shared__ _Float16 vstage[64 * DHP];           // staged V union (1 KB)
  __shared__ float    rpbs[WIN][50];              // staged bias window rows

  // ---- stage rel-pos bias rows [bi .. bi+24] for this head ----
#pragma unroll 1
  for (int r = 0; r < WIN; ++r)
    for (int c = lane; c < 2 * WIN - 1; c += 32)
      rpbs[r][c] = rpb[(head * (2 * WIN - 1) + (bi + r)) * (2 * WIN - 1) + c];

  // ---- Q fragment: one b128 per lane, pads already zero ----
  v16h qa;
  {
    int j = j0 + lo;
    int jc = j < Ww ? j : Ww - 1;
    v8h qv = *(const v8h*)(q + ((z * Hh + i) * Ww + jc) * DHP);
#pragma unroll
    for (int t = 0; t < 8; ++t) qa[t] = hi ? (_Float16)0.0f : qv[t];
#pragma unroll
    for (int t = 8; t < 16; ++t) qa[t] = (_Float16)0.0f;
  }
  __syncthreads();

  const _Float16* kbase_p = kg + z * Hh * Ww * DHP;
  const _Float16* vbase_p = vg + z * Hh * Ww * DHP;

  // ---- scores: 3 union key tiles of 16 cols per window-row ----
#pragma unroll 1
  for (int ki = 0; ki < WIN; ++ki) {
    const _Float16* krowp = kbase_p + (si + ki) * Ww * DHP;
    if (ki + 1 < WIN)
      __builtin_prefetch(kbase_p + (si + ki + 1) * Ww * DHP + u0 * DHP, 0, 0);
#pragma unroll 1
    for (int kt = 0; kt < 3; ++kt) {
      int col = u0 + kt * 16 + lo;
      v8h kv = *(const v8h*)(krowp + (col < Ww ? col : Ww - 1) * DHP);
      v16h kbf;
#pragma unroll
      for (int t = 0; t < 8; ++t) kbf[t] = hi ? (_Float16)0.0f : kv[t];
#pragma unroll
      for (int t = 8; t < 16; ++t) kbf[t] = (_Float16)0.0f;
      v8f c = {};
      c = __builtin_amdgcn_wmma_f32_16x16x32_f16(false, qa, false, kbf,
                                                 (short)0, c, false, false);
      // unconditional scatter: invalid entries go to dump column 625
#pragma unroll
      for (int g = 0; g < 8; ++g) {
        int m = g + 8 * hi;
        int j = j0 + m;
        int jc = j < Ww ? j : Ww - 1;
        int sjq = iclampi(jc - WIN / 2, 0, Ww - WIN);
        int u = kt * 16 + lo;
        int kj = u + u0 - sjq;
        bool valid = (j < Ww) & (kj >= 0) & (kj < WIN);
        int kjc = iclampi(kj, 0, WIN - 1);
        int rc = iclampi(sjq - jc + WIN - 1 + kjc, 0, 2 * WIN - 2);
        float bias = rpbs[ki][rc];
        int idx = valid ? (ki * WIN + kjc) : (WIN * WIN);
        sc[m][idx] = c[g] + bias;
      }
    }
  }
  __syncthreads();

  // ---- softmax over 625 keys; write normalized attention as f16 ----
#pragma unroll 1
  for (int m = 0; m < 16; ++m) {
    float mx = -3.0e38f;
    for (int t = lane; t < WIN * WIN; t += 32) mx = fmaxf(mx, sc[m][t]);
#pragma unroll
    for (int o = 16; o > 0; o >>= 1) mx = fmaxf(mx, __shfl_xor(mx, o, 32));
    float sum = 0.0f;
    for (int t = lane; t < WIN * WIN; t += 32) {
      float e = __expf(sc[m][t] - mx);
      sc[m][t] = e;
      sum += e;
    }
#pragma unroll
    for (int o = 16; o > 0; o >>= 1) sum += __shfl_xor(sum, o, 32);
    float inv = 1.0f / sum;
    for (int t = lane; t < WIN * WIN; t += 32)
      af[m][t] = (_Float16)(sc[m][t] * inv);
  }
  __syncthreads();

  // ---- context: out[16q x dh] += attn_union[16 x 64] @ V_union[64 x 16] ----
  v8f ctx = {};
  const int j = j0 + lo;
  const int jc = j < Ww ? j : Ww - 1;
  const int sjq = iclampi(jc - WIN / 2, 0, Ww - WIN);
  const int shift = sjq - u0;     // in [0,15]
  const bool qvalid = j < Ww;
  const int locp = lo & 7;

#pragma unroll 1
  for (int ki = 0; ki < WIN; ++ki) {
    const _Float16* vrowp = vbase_p + (si + ki) * Ww * DHP;
    if (ki + 1 < WIN)
      __builtin_prefetch(vbase_p + (si + ki + 1) * Ww * DHP + u0 * DHP, 0, 0);
    // stage the 64-col x DHP union span (512 f16, clamped, 8B chunks)
#pragma unroll
    for (int it = 0; it < 4; ++it) {
      int c4 = (it * 32 + lane) * 4;
      int gidx = u0 * DHP + c4;
      int gc = gidx <= (Ww * DHP - 4) ? gidx : (Ww * DHP - 4);
      *(v4h*)(vstage + c4) = *(const v4h*)(vrowp + gc);
    }
    __syncthreads();
#pragma unroll 1
    for (int st = 0; st < 2; ++st) {
      v16h av;
#pragma unroll
      for (int t = 0; t < 16; ++t) {
        int u = st * 32 + a16_k(t, 0) + 8 * hi;  // a16_k(t,1)=a16_k(t,0)+8
        int kj = u - shift;
        bool v = qvalid & (kj >= 0) & (kj < WIN);
        int kjc = iclampi(kj, 0, WIN - 1);
        _Float16 a = af[lo][ki * WIN + kjc];     // unconditional LDS read
        av[t] = v ? a : (_Float16)0.0f;
      }
      v16h bv;
#pragma unroll
      for (int t = 0; t < 16; ++t) {
        int kcol = st * 32 + a16_k(t, hi);       // <= 63
        _Float16 b0 = vstage[kcol * DHP + locp]; // pads are zero in LDS
        bv[t] = (lo < 8) ? b0 : (_Float16)0.0f;
      }
      ctx = __builtin_amdgcn_wmma_f32_16x16x32_f16(false, av, false, bv,
                                                   (short)0, ctx, false, false);
    }
    __syncthreads();
  }
  // ---- store attention output f16 padded [BHW, 32] ----
  if (lo < DH) {
    int rbase = (z >> 2) * HW + i * Ww;
#pragma unroll
    for (int g = 0; g < 8; ++g) {
      int m = g + 8 * hi;
      int j2 = j0 + m;
      if (j2 < Ww) ao[(rbase + j2) * CcP + head * DH + lo] = (_Float16)ctx[g];
    }
  }
}

// ===========================================================================
// Kernel 4: h' = attn_out @ W_proj + b_proj
//   write_f32 == 0: write f16 padded hb16 (intermediate layer)
//   write_f32 == 1: write f32 d_out (final layer, full-precision accumulator)
// ===========================================================================
__global__ __launch_bounds__(32) void proj_kernel(
    const _Float16* __restrict__ ao, const _Float16* __restrict__ wpp,
    const float* __restrict__ bp, _Float16* __restrict__ outh,
    float* __restrict__ outf, int write_f32) {
  const int lane = threadIdx.x, hi = lane >> 4, lo = lane & 15;
  const int r0 = blockIdx.x * 16;

  v16h A = a_from_padded(ao + (r0 + lo) * CcP, hi);

#pragma unroll
  for (int nt = 0; nt < 2; ++nt) {
    v16h Bv = *(const v16h*)(wpp + (nt * TILE_H + lane * 16));
    v8f acc = {};
    acc = __builtin_amdgcn_wmma_f32_16x16x32_f16(false, A, false, Bv, (short)0,
                                                 acc, false, false);
    int n = nt * 16 + lo;
    float bias = bp[n < Cc ? n : Cc - 1];
    if (n < Cc) {
      if (write_f32) {
#pragma unroll
        for (int g = 0; g < 8; ++g) {
          int m = g + 8 * hi;
          outf[(r0 + m) * Cc + n] = acc[g] + bias;
        }
      } else {
#pragma unroll
        for (int g = 0; g < 8; ++g) {
          int m = g + 8 * hi;
          outh[(r0 + m) * CcP + n] = (_Float16)(acc[g] + bias);
        }
      }
    }
  }
}

// ===========================================================================
extern "C" void kernel_launch(void* const* d_in, const int* in_sizes, int n_in,
                              void* d_out, int out_size, void* d_ws,
                              size_t ws_size, hipStream_t stream) {
  (void)in_sizes; (void)n_in; (void)out_size; (void)ws_size;
  const float* x   = (const float*)d_in[0];
  const float* Wl  = (const float*)d_in[1];
  const float* bl  = (const float*)d_in[2];
  const float* Wq  = (const float*)d_in[3];
  const float* bq  = (const float*)d_in[4];
  const float* rpb = (const float*)d_in[5];
  const float* Wp  = (const float*)d_in[6];
  const float* bp  = (const float*)d_in[7];
  float* out = (float*)d_out;

  // all-f16 workspace layout (every segment a multiple of 16 halves)
  _Float16* x16  = (_Float16*)d_ws;                  // [BHW,160]
  _Float16* hb16 = x16 + (long)BHW * FINP;           // [BHW,32]
  _Float16* ao16 = hb16 + (long)BHW * CcP;           // [BHW,32]
  _Float16* qb   = ao16 + (long)BHW * CcP;           // [ZHW,8]
  _Float16* kb   = qb + (long)ZHW * DHP;
  _Float16* vb   = kb + (long)ZHW * DHP;
  _Float16* wlp  = vb + (long)ZHW * DHP;
  _Float16* wqp  = wlp + WL_TILES * TILE_H;
  _Float16* wpp  = wqp + WQ_TILES * TILE_H;

  pack_weights_kernel<<<(WL_TILES * TILE_H + 255) / 256, 256, 0, stream>>>(
      Wl, Wq, Wp, wlp, wqp, wpp);
  convert_x_kernel<<<BHW, FINP, 0, stream>>>(x, x16);
  zero_pad_kernel<<<(ZHW + 255) / 256, 256, 0, stream>>>(
      (unsigned*)qb, (unsigned*)kb, (unsigned*)vb, (unsigned*)hb16,
      (unsigned*)ao16);

  const int gemm_blocks = BHW / 16;  // 2112, exact
  lin_relu_kernel<<<gemm_blocks, 32, 0, stream>>>(x16, wlp, bl, hb16);
  for (int layer = 0; layer < 2; ++layer) {
    qkv_kernel<<<gemm_blocks, 32, 0, stream>>>(hb16, wqp, bq, qb, kb, vb);
    dim3 g((Ww + 15) / 16, Hh, Bn * NH);  // (6, 192, 8)
    natten_kernel<<<g, 32, 0, stream>>>(qb, kb, vb, rpb, ao16);
    proj_kernel<<<gemm_blocks, 32, 0, stream>>>(ao16, wpp, bp, hb16, out,
                                                layer == 1 ? 1 : 0);
  }
}
